// PlainMamba2D_2473901163054
// MI455X (gfx1250) — compile-verified, matrix operand order checked by reading, not run on hardware
//
#include <hip/hip_runtime.h>
#include <math.h>

typedef __attribute__((ext_vector_type(16))) _Float16 v16h;
typedef __attribute__((ext_vector_type(8)))  float    v8f;
typedef __attribute__((ext_vector_type(4)))  float    v4f;

#define BATCH 4
#define HH 32
#define WW 32
#define LL (HH*WW)            // 1024
#define DMODEL 384
#define EE 768                // D_INNER
#define NST 16                // D_STATE
#define RNK 24                // DT_RANK
#define XPROJ (RNK + 2*NST)   // 56 (padded to 64 in ws)
#define XPROJ_P 64
#define CONVK 7

// Pack two contiguous 4-float runs into 8 consecutive f16 elements of a WMMA operand.
__device__ __forceinline__ void pack8(v16h& dst, int base, v4f lo, v4f hi) {
#pragma unroll
  for (int i = 0; i < 4; ++i) {
    dst[base + i]     = (_Float16)lo[i];
    dst[base + 4 + i] = (_Float16)hi[i];
  }
}

// C[M,N] = epilogue( A[M,K] @ Wt[N,K]^T ).  One wave per 16x32 output tile
// (2 WMMAs per K-slice sharing one A fetch).  Requires K%32==0, N%32==0,
// rows 16B-aligned.  Per the ISA 16-bit operand layout, each lane's 16 packed
// f16s are two contiguous 8-float runs: [k0+8*half, +8) and [k0+16+8*half, +8).
// MODE 0: plain store; MODE 1: softplus(acc + 2*bias[n]).
template<int MODE>
__global__ void wmma_gemm_kernel(const float* __restrict__ A, int lda,
                                 const float* __restrict__ Wt, int ldw,
                                 float* __restrict__ C, int ldc,
                                 int K, const float* __restrict__ bias)
{
  int n0   = blockIdx.x * 32;
  int m0   = blockIdx.y * 16;
  int lane = threadIdx.x & 31;   // wave32
  int half = lane >> 4;
  int lm   = lane & 15;
  const float* Ar  = A  + (size_t)(m0 + lm) * lda;
  const float* Br0 = Wt + (size_t)(n0 + lm) * ldw;
  const float* Br1 = Wt + (size_t)(n0 + 16 + lm) * ldw;
  v8f acc0 = {}, acc1 = {};
  for (int k0 = 0; k0 < K; k0 += 32) {
    int q = k0 + 8 * half;
    v16h a, b0, b1;
    pack8(a,  0, *(const v4f*)(Ar  + q),      *(const v4f*)(Ar  + q + 4));
    pack8(a,  8, *(const v4f*)(Ar  + q + 16), *(const v4f*)(Ar  + q + 20));
    pack8(b0, 0, *(const v4f*)(Br0 + q),      *(const v4f*)(Br0 + q + 4));
    pack8(b0, 8, *(const v4f*)(Br0 + q + 16), *(const v4f*)(Br0 + q + 20));
    pack8(b1, 0, *(const v4f*)(Br1 + q),      *(const v4f*)(Br1 + q + 4));
    pack8(b1, 8, *(const v4f*)(Br1 + q + 16), *(const v4f*)(Br1 + q + 20));
    acc0 = __builtin_amdgcn_wmma_f32_16x16x32_f16(
        false, a, false, b0, (short)0, acc0, false, false);
    acc1 = __builtin_amdgcn_wmma_f32_16x16x32_f16(
        false, a, false, b1, (short)0, acc1, false, false);
  }
  int n0l = n0 + lm, n1l = n0 + 16 + lm;
#pragma unroll
  for (int v = 0; v < 8; ++v) {
    int mm = m0 + half * 8 + v;            // D layout: VGPR v -> M = v + 8*half
    float r0 = acc0[v], r1 = acc1[v];
    if (MODE == 1) {
      float x0 = r0 + 2.0f * bias[n0l];
      float x1 = r1 + 2.0f * bias[n1l];
      r0 = (x0 > 20.0f) ? x0 : log1pf(__expf(x0));
      r1 = (x1 > 20.0f) ? x1 : log1pf(__expf(x1));
    }
    C[(size_t)mm * ldc + n0l] = r0;
    C[(size_t)mm * ldc + n1l] = r1;
  }
}

// Zero-pad W_xproj (56x768 -> 64x768, zero rows).
__global__ void pad_wxproj_kernel(const float* __restrict__ Wxp, float* __restrict__ out) {
  int idx = blockIdx.x * 256 + threadIdx.x;
  if (idx >= 64 * EE) return;
  int n = idx / EE, k = idx - n * EE;
  out[idx] = (n < XPROJ) ? Wxp[n * EE + k] : 0.0f;
}

// Zero-pad W_dt (768x24 -> 768x32, zero cols).
__global__ void pad_wdt_kernel(const float* __restrict__ Wdt, float* __restrict__ out) {
  int idx = blockIdx.x * 256 + threadIdx.x;
  if (idx >= EE * 32) return;
  int n = idx >> 5, k = idx & 31;
  out[idx] = (k < RNK) ? Wdt[n * RNK + k] : 0.0f;
}

// Exact device-side replica of get_permute_order (orders + step-dirs).
__global__ void build_orders_kernel(int* __restrict__ orders, int* __restrict__ dirs)
{
  if (threadIdx.x != 0 || blockIdx.x != 0) return;
  const int H = HH, W = WW, L = LL;
  { int i = 0, j = 0, right = 1, idx = 0;          // 1: row snake from (0,0)
    while (i < H) {
      orders[0*L + idx] = i*W + j; int d;
      if (right) { if (j < W-1) { j++; d = 1; } else { i++; d = 4; right = 0; } }
      else       { if (j > 0)   { j--; d = 2; } else { i++; d = 4; right = 1; } }
      if (idx + 1 < L) dirs[0*L + idx + 1] = d;
      idx++;
    }
    dirs[0*L] = 0;
  }
  { int i, j, right;                                // 2: row snake from bottom
    if (H & 1) { i = H-1; j = W-1; right = 0; } else { i = H-1; j = 0; right = 1; }
    int idx = 0;
    while (i > -1) {
      orders[1*L + idx] = i*W + j; int d;
      if (right) { if (j < W-1) { j++; d = 1; } else { i--; d = 3; right = 0; } }
      else       { if (j > 0)   { j--; d = 2; } else { i--; d = 3; right = 1; } }
      if (idx + 1 < L) dirs[1*L + idx + 1] = d;
      idx++;
    }
    dirs[1*L] = 0;
  }
  { int i = 0, j = 0, down = 1, idx = 0;            // 3: column snake from (0,0)
    while (j < W) {
      orders[2*L + idx] = i*W + j; int d;
      if (down) { if (i < H-1) { i++; d = 4; } else { j++; d = 1; down = 0; } }
      else      { if (i > 0)   { i--; d = 3; } else { j++; d = 1; down = 1; } }
      if (idx + 1 < L) dirs[2*L + idx + 1] = d;
      idx++;
    }
    dirs[2*L] = 0;
  }
  { int i, j, down;                                 // 4: column snake from right
    if (W & 1) { i = H-1; j = W-1; down = 0; } else { i = 0; j = W-1; down = 1; }
    int idx = 0;
    while (j > -1) {
      orders[3*L + idx] = i*W + j; int d;
      if (down) { if (i < H-1) { i++; d = 4; } else { j--; d = 2; down = 0; } }
      else      { if (i > 0)   { i--; d = 3; } else { j--; d = 2; down = 1; } }
      if (idx + 1 < L) dirs[3*L + idx + 1] = d;
      idx++;
    }
    dirs[3*L] = 0;
  }
}

// Depthwise 7x7 conv (pad 3) + bias + SiLU, reading the xx half of xz.
__global__ void conv_silu_kernel(const float* __restrict__ xz,
                                 const float* __restrict__ conv_w,
                                 const float* __restrict__ conv_b,
                                 float* __restrict__ x_conv)
{
  int bl = blockIdx.x;                  // b*L + l
  int b  = bl / LL;
  int l  = bl - b * LL;
  int i  = l / WW, j = l - i * WW;
  for (int e = threadIdx.x; e < EE; e += blockDim.x) {
    float s = conv_b[e];
#pragma unroll
    for (int di = 0; di < CONVK; ++di) {
      int ii = i + di - CONVK/2;
      if (ii < 0 || ii >= HH) continue;
#pragma unroll
      for (int dj = 0; dj < CONVK; ++dj) {
        int jj = j + dj - CONVK/2;
        if (jj < 0 || jj >= WW) continue;
        s += xz[((size_t)(b*LL + ii*WW + jj))*(2*EE) + e] * conv_w[e*49 + di*7 + dj];
      }
    }
    x_conv[(size_t)bl * EE + e] = s / (1.0f + __expf(-s));
  }
}

// Selective scan: one block per (direction k, batch b), one thread per channel e.
__global__ void scan_kernel(const float* __restrict__ x_conv,
                            const float* __restrict__ x_dbl,
                            const float* __restrict__ delta,
                            const float* __restrict__ A_log,
                            const float* __restrict__ dirB,
                            const float* __restrict__ Dp,
                            const int* __restrict__ orders,
                            const int* __restrict__ dirs,
                            float* __restrict__ ys)
{
  int k = blockIdx.x >> 2;
  int b = blockIdx.x & 3;
  int e = threadIdx.x;                  // blockDim = 768
  float Aval[NST], h[NST];
#pragma unroll
  for (int n = 0; n < NST; ++n) {
    Aval[n] = -__expf(A_log[e*NST + n]);
    h[n] = 0.0f;
  }
  float De = Dp[e];
  float* ysk = ys + (size_t)(k*BATCH + b) * LL * EE;
  for (int t = 0; t < LL; ++t) {
    int l  = orders[k*LL + t];          // only u is permuted (reference quirk)
    int dr = dirs[k*LL + t];
    float d = delta[((size_t)b*LL + t)*EE + e];
    float u = x_conv[((size_t)b*LL + l)*EE + e];
    const float* Bm = x_dbl + ((size_t)b*LL + t)*XPROJ_P + RNK;
    const float* Cm = Bm + NST;
    const float* dB = dirB + dr*NST;
    float du = d * u;
    float y = 0.0f;
#pragma unroll
    for (int n = 0; n < NST; ++n) {
      float dA = __expf(d * Aval[n]);
      h[n] = dA * h[n] + du * (Bm[n] + dB[n]);
      y += h[n] * Cm[n];
    }
    ysk[(size_t)l * EE + e] = y + u * De;   // scatter == inverse-permute
  }
}

// y_sum over 4 directions, gated by silu(z) (z half of xz).
__global__ void gate_kernel(const float* __restrict__ ys,
                            const float* __restrict__ xz,
                            float* __restrict__ g)
{
  size_t idx = (size_t)blockIdx.x * blockDim.x + threadIdx.x;
  if (idx >= (size_t)BATCH*LL*EE) return;
  int e = (int)(idx % EE);
  size_t bl = idx / EE;
  int l = (int)(bl % LL);
  int b = (int)(bl / LL);
  float s = 0.0f;
#pragma unroll
  for (int k = 0; k < 4; ++k)
    s += ys[(((size_t)(k*BATCH + b))*LL + l)*EE + e];
  float z = xz[bl*(2*EE) + EE + e];
  g[idx] = s * (z / (1.0f + __expf(-z)));
}

extern "C" void kernel_launch(void* const* d_in, const int* in_sizes, int n_in,
                              void* d_out, int out_size, void* d_ws, size_t ws_size,
                              hipStream_t stream) {
  (void)in_sizes; (void)n_in; (void)out_size; (void)ws_size;
  const float* x       = (const float*)d_in[0];
  // d_in[1]=H, d_in[2]=W (compile-time constants 32x32 here)
  const float* W_in    = (const float*)d_in[3];
  const float* conv_w  = (const float*)d_in[4];
  const float* conv_b  = (const float*)d_in[5];
  const float* W_xproj = (const float*)d_in[6];
  const float* W_dt    = (const float*)d_in[7];
  const float* b_dt    = (const float*)d_in[8];
  const float* A_log   = (const float*)d_in[9];
  const float* Dp      = (const float*)d_in[10];
  const float* W_out   = (const float*)d_in[11];
  const float* dirB    = (const float*)d_in[12];
  float* out = (float*)d_out;

  // workspace layout (floats); every byte read is written earlier this launch
  float* ws = (float*)d_ws;
  size_t off = 0;
  float* xz      = ws + off; off += (size_t)BATCH*LL*2*EE;     // 4096 x 1536
  float* x_conv  = ws + off; off += (size_t)BATCH*LL*EE;       // 4096 x 768
  float* x_dbl   = ws + off; off += (size_t)BATCH*LL*XPROJ_P;  // 4096 x 64 (padded)
  float* delta   = ws + off; off += (size_t)BATCH*LL*EE;       // 4096 x 768
  float* ysb     = ws + off; off += (size_t)4*BATCH*LL*EE;     // 4 x 4096 x 768
  float* g       = ws + off; off += (size_t)BATCH*LL*EE;       // 4096 x 768
  float* Wxp_pad = ws + off; off += (size_t)64*EE;             // 64 x 768
  float* Wdt_pad = ws + off; off += (size_t)EE*32;             // 768 x 32
  int* orders = (int*)(ws + off);
  int* dirs   = orders + 4*LL;

  build_orders_kernel<<<1, 32, 0, stream>>>(orders, dirs);
  pad_wxproj_kernel<<<(64*EE + 255)/256, 256, 0, stream>>>(W_xproj, Wxp_pad);
  pad_wdt_kernel<<<(EE*32 + 255)/256, 256, 0, stream>>>(W_dt, Wdt_pad);

  dim3 wave(32);
  // 1) xz = x @ W_in^T             (M=4096, K=384, N=1536)
  wmma_gemm_kernel<0><<<dim3((2*EE)/32, (BATCH*LL)/16), wave, 0, stream>>>(
      x, DMODEL, W_in, DMODEL, xz, 2*EE, DMODEL, nullptr);
  // 2) depthwise conv + SiLU
  conv_silu_kernel<<<BATCH*LL, 256, 0, stream>>>(xz, conv_w, conv_b, x_conv);
  // 3) x_dbl = x_conv @ W_xproj_pad^T  (M=4096, K=768, N=64 padded)
  wmma_gemm_kernel<0><<<dim3(XPROJ_P/32, (BATCH*LL)/16), wave, 0, stream>>>(
      x_conv, EE, Wxp_pad, EE, x_dbl, XPROJ_P, EE, nullptr);
  // 4) delta = softplus(dt_low @ W_dt_pad^T + 2*b_dt)  (M=4096, K=32 padded, N=768)
  //    A cols 24..31 hold Bm values but meet zero weights -> exact.
  wmma_gemm_kernel<1><<<dim3(EE/32, (BATCH*LL)/16), wave, 0, stream>>>(
      x_dbl, XPROJ_P, Wdt_pad, 32, delta, EE, 32, b_dt);
  // 5) 4-direction selective scan (16 blocks of 768 threads)
  scan_kernel<<<4*BATCH, EE, 0, stream>>>(x_conv, x_dbl, delta, A_log, dirB, Dp,
                                          orders, dirs, ysb);
  // 6) sum over directions, gate with silu(z)
  size_t tot = (size_t)BATCH*LL*EE;
  gate_kernel<<<(unsigned)((tot + 255)/256), 256, 0, stream>>>(ysb, xz, g);
  // 7) out = g @ W_out^T           (M=4096, K=768, N=384)
  wmma_gemm_kernel<0><<<dim3(DMODEL/32, (BATCH*LL)/16), wave, 0, stream>>>(
      g, EE, W_out, EE, out, DMODEL, EE, nullptr);
}
